// AutoregressiveGaussian_5592047419611
// MI455X (gfx1250) — matrix operand order repaired and checked by reading, not compiled, not run on hardware
//
#include <hip/hip_runtime.h>
#include <hip/hip_bf16.h>
#include <math.h>

typedef __attribute__((ext_vector_type(16))) __bf16 v16bf;
typedef __attribute__((ext_vector_type(8)))  __bf16 v8bf;
typedef __attribute__((ext_vector_type(8)))  float  v8f;

constexpr int kF = 128;           // feature dim
constexpr int kD = 256;           // inner depth (2F)
constexpr int kG = 1024;          // 4*D gate width
constexpr int kS = 512;           // sequence length
constexpr int kB = 4096;          // batch
constexpr int kRows = 32;         // batch rows per workgroup (2 M-tiles)
constexpr int kThreads = 256;     // 8 waves (wave32)

static __device__ __forceinline__ v8f vfill(float x) {
  v8f r;
#pragma unroll
  for (int i = 0; i < 8; ++i) r[i] = x;
  return r;
}

static __device__ __forceinline__ v8f wmma_bf16(v16bf a, v16bf b, v8f c) {
  // v_wmma_f32_16x16x32_bf16: D = A(16x32) * B(32x16) + C
  return __builtin_amdgcn_wmma_f32_16x16x32_bf16(false, a, false, b, (short)0, c,
                                                 false, false);
}

// A fragment (16x32 bf16 tile from row-major [rows x ldk] buffer):
//   lane l<16  -> row l,     K = k..k+15
//   lane l>=16 -> row l-16,  K = k+16..k+31
static __device__ __forceinline__ v16bf load_a(const __bf16* base, int row0,
                                               int ldk, int k, int lane) {
  const __bf16* p = base + (row0 + (lane & 15)) * ldk + k + ((lane >> 4) << 4);
  v8bf lo = *(const v8bf*)p;
  v8bf hi = *(const v8bf*)(p + 8);
  return __builtin_shufflevector(lo, hi, 0, 1, 2, 3, 4, 5, 6, 7, 8, 9, 10, 11,
                                 12, 13, 14, 15);
}

// B fragment (32x16 bf16): B = W^T tile, B[k][n] = W[nbase+n][k].
//   lane l<16  -> column n=l,    K = k..k+15   (one contiguous row chunk of W)
//   lane l>=16 -> column n=l-16, K = k+16..k+31
static __device__ __forceinline__ v16bf load_b(const __bf16* __restrict__ W,
                                               int nbase, int ldk, int k,
                                               int lane) {
  return *(const v16bf*)(W + (nbase + (lane & 15)) * ldk + k +
                         ((lane >> 4) << 4));
}

static __device__ __forceinline__ float sigm_(float x) {
  return 1.0f / (1.0f + __expf(-x));
}
static __device__ __forceinline__ float gelu_(float x) {
  return 0.5f * x * (1.0f + erff(x * 0.70710678118654752f));
}

// ---------------------------------------------------------------------------
// Kernel 0: cast weights to bf16 (L2-resident), fuse b = b_ih + b_hh.
// ---------------------------------------------------------------------------
__global__ void prep_kernel(const float* __restrict__ wih,
                            const float* __restrict__ whh,
                            const float* __restrict__ outw,
                            const float* __restrict__ zw,
                            const float* __restrict__ bih,
                            const float* __restrict__ bhh,
                            __bf16* __restrict__ wihb, __bf16* __restrict__ whhb,
                            __bf16* __restrict__ outwb, __bf16* __restrict__ zwb,
                            float* __restrict__ bcomb) {
  int i = blockIdx.x * blockDim.x + threadIdx.x;
  if (i < kG * kF) wihb[i] = (__bf16)wih[i];
  if (i < kG * kD) whhb[i] = (__bf16)whh[i];
  if (i < kF * kD) {
    outwb[i] = (__bf16)outw[i];
    zwb[i] = (__bf16)zw[i];
  }
  if (i < kG) bcomb[i] = bih[i] + bhh[i];
}

// ---------------------------------------------------------------------------
// Kernel 1: preamble MLP (Linear-GELU x2 -> Linear), hidden/cell init.
// One block per batch row; tiny (0.4 GFLOP total), VALU is fine.
// ---------------------------------------------------------------------------
__global__ void __launch_bounds__(kF) preamble_kernel(
    const float* __restrict__ noise, const float* __restrict__ w1,
    const float* __restrict__ b1, const float* __restrict__ w2,
    const float* __restrict__ b2, const float* __restrict__ w3,
    const float* __restrict__ b3, const float* __restrict__ hw,
    const float* __restrict__ hb, float* __restrict__ x0,
    float* __restrict__ h0, float* __restrict__ c0) {
  __shared__ float sa[kF], sb[kF];
  const int b = blockIdx.x, f = threadIdx.x;
  sa[f] = noise[b * kF + f];
  __syncthreads();
  float acc = b1[f];
  for (int k = 0; k < kF; ++k) acc = fmaf(sa[k], w1[f * kF + k], acc);
  sb[f] = gelu_(acc);
  __syncthreads();
  acc = b2[f];
  for (int k = 0; k < kF; ++k) acc = fmaf(sb[k], w2[f * kF + k], acc);
  sa[f] = gelu_(acc);
  __syncthreads();
  acc = b3[f];
  for (int k = 0; k < kF; ++k) acc = fmaf(sa[k], w3[f * kF + k], acc);
  x0[b * kF + f] = acc;
  sb[f] = acc;
  __syncthreads();
  float hA = hb[f], hB = hb[f + kF];
  for (int k = 0; k < kF; ++k) {
    hA = fmaf(sb[k], hw[f * kF + k], hA);
    hB = fmaf(sb[k], hw[(f + kF) * kF + k], hB);
  }
  h0[b * kD + f] = hA;
  h0[b * kD + f + kF] = hB;
  c0[b * kD + f] = tanhf(hA);
  c0[b * kD + f + kF] = tanhf(hB);
}

// ---------------------------------------------------------------------------
// Kernel 2: persistent per-row-tile sequence kernel. 128 WGs x 8 waves.
// Each WG owns 32 batch rows for all 512 steps (no inter-WG sync needed).
// Wave w owns D-columns [32w, 32w+32) of the LSTM state.
// Cell state c stays in VGPRs for the whole sequence.
// Weight base pointers are laundered through an opaque SGPR zero each step so
// LICM cannot hoist the 224 B-fragments out of the t-loop (which previously
// spilled ~1800 VGPRs to scratch); weights stream from L2 every step instead.
// ---------------------------------------------------------------------------
__global__ void __launch_bounds__(kThreads, 1) seq_kernel(
    const __bf16* __restrict__ wih_g, const __bf16* __restrict__ whh_g,
    const __bf16* __restrict__ outw_g, const __bf16* __restrict__ zw_g,
    const float* __restrict__ bcomb, const float* __restrict__ outb,
    const float* __restrict__ zb, const float* __restrict__ x0,
    const float* __restrict__ h0, const float* __restrict__ c0,
    const float* __restrict__ eps, float* __restrict__ out) {
  __shared__ __bf16 lds_x[kRows * kF];       // current input sample (bf16)
  __shared__ __bf16 lds_h[2][kRows * kD];    // double-buffered hidden (bf16)
  __shared__ __bf16 lds_g[kRows * kF];       // gelu(out) (bf16)

  const int tid = threadIdx.x;
  const int wave = tid >> 5;
  const int lane = tid & 31;
  const int l15 = lane & 15;
  const int hshift = (lane >> 4) << 3;  // +8 rows for lanes 16-31 (C/D layout)
  const int rowbase = blockIdx.x * kRows;
  const int dlo = wave * 32;            // this wave's D-column slice

  // ---- prologue: stage x0/h0, pull c0 into registers ----
  for (int i = tid; i < kRows * kF; i += kThreads)
    lds_x[i] = (__bf16)x0[rowbase * kF + i];
  for (int i = tid; i < kRows * kD; i += kThreads)
    lds_h[0][i] = (__bf16)h0[rowbase * kD + i];

  float c_reg[32];
#pragma unroll
  for (int dn = 0; dn < 2; ++dn)
#pragma unroll
    for (int mt = 0; mt < 2; ++mt)
#pragma unroll
      for (int v = 0; v < 8; ++v) {
        int row = mt * 16 + hshift + v;
        int col = dlo + dn * 16 + l15;
        c_reg[(dn * 2 + mt) * 8 + v] = c0[(size_t)(rowbase + row) * kD + col];
      }

  // per-lane constant biases (hoisted out of the t-loop)
  float bi[2], bff[2], bg[2], bo[2];
#pragma unroll
  for (int dn = 0; dn < 2; ++dn) {
    int col = dlo + dn * 16 + l15;
    bi[dn] = bcomb[0 * kD + col];
    bff[dn] = bcomb[1 * kD + col];
    bg[dn] = bcomb[2 * kD + col];
    bo[dn] = bcomb[3 * kD + col];
  }
  const float bout = outb[wave * 16 + l15];
  const float bmu = zb[wave * 16 + l15];
  const float bls = zb[kF + wave * 16 + l15];

  __syncthreads();

  int opq = 0;  // opaque zero (SGPR), redefined every step to defeat LICM

  for (int t = 0; t < kS; ++t) {
    asm volatile("" : "+s"(opq));
    const __bf16* wih = wih_g + opq;
    const __bf16* whh = whh_g + opq;
    const __bf16* outw = outw_g + opq;
    const __bf16* zw = zw_g + opq;

    const __bf16* hold = lds_h[t & 1];
    __bf16* hnew = lds_h[(t + 1) & 1];

    // ---- phase 1: gates = x@Wih^T + h@Whh^T + b; LSTM update ----
#pragma unroll
    for (int dn = 0; dn < 2; ++dn) {
      const int ncol = dlo + dn * 16;
#pragma unroll
      for (int mt = 0; mt < 2; ++mt) {
        v8f ai = vfill(bi[dn]), af = vfill(bff[dn]);
        v8f ag = vfill(bg[dn]), ao = vfill(bo[dn]);
#pragma unroll
        for (int k = 0; k < kF; k += 32) {
          v16bf a = load_a(lds_x, mt * 16, kF, k, lane);
          ai = wmma_bf16(a, load_b(wih, 0 * kD + ncol, kF, k, lane), ai);
          af = wmma_bf16(a, load_b(wih, 1 * kD + ncol, kF, k, lane), af);
          ag = wmma_bf16(a, load_b(wih, 2 * kD + ncol, kF, k, lane), ag);
          ao = wmma_bf16(a, load_b(wih, 3 * kD + ncol, kF, k, lane), ao);
        }
#pragma unroll
        for (int k = 0; k < kD; k += 32) {
          v16bf a = load_a(hold, mt * 16, kD, k, lane);
          ai = wmma_bf16(a, load_b(whh, 0 * kD + ncol, kD, k, lane), ai);
          af = wmma_bf16(a, load_b(whh, 1 * kD + ncol, kD, k, lane), af);
          ag = wmma_bf16(a, load_b(whh, 2 * kD + ncol, kD, k, lane), ag);
          ao = wmma_bf16(a, load_b(whh, 3 * kD + ncol, kD, k, lane), ao);
        }
        const int colr = ncol + l15;
        const int rb = mt * 16 + hshift;
#pragma unroll
        for (int v = 0; v < 8; ++v) {
          int ci = (dn * 2 + mt) * 8 + v;
          float c = sigm_(af[v]) * c_reg[ci] + sigm_(ai[v]) * tanhf(ag[v]);
          c_reg[ci] = c;
          float h = sigm_(ao[v]) * tanhf(c);
          hnew[(rb + v) * kD + colr] = (__bf16)h;
        }
      }
    }
    __syncthreads();

    // ---- phase 2: g = gelu(h @ out_w^T + out_b); wave owns 16 out-cols ----
    {
      const int ncol = wave * 16;
#pragma unroll
      for (int mt = 0; mt < 2; ++mt) {
        v8f acc = vfill(bout);
#pragma unroll
        for (int k = 0; k < kD; k += 32) {
          v16bf a = load_a(hnew, mt * 16, kD, k, lane);
          acc = wmma_bf16(a, load_b(outw, ncol, kD, k, lane), acc);
        }
        const int colr = ncol + l15;
        const int rb = mt * 16 + hshift;
#pragma unroll
        for (int v = 0; v < 8; ++v)
          lds_g[(rb + v) * kF + colr] = (__bf16)gelu_(acc[v]);
      }
    }
    __syncthreads();

    // ---- phase 3: (mu,ls) = g @ z_w^T + z_b; sample = mu + exp(ls)*eps ----
    {
      const int ncol = wave * 16;  // mu cols; logsigma cols = ncol + 128
#pragma unroll
      for (int mt = 0; mt < 2; ++mt) {
        v8f amu = vfill(bmu), als = vfill(bls);
#pragma unroll
        for (int k = 0; k < kF; k += 32) {
          v16bf a = load_a(lds_g, mt * 16, kF, k, lane);
          amu = wmma_bf16(a, load_b(zw, ncol, kF, k, lane), amu);
          als = wmma_bf16(a, load_b(zw, kF + ncol, kF, k, lane), als);
        }
        const int colr = ncol + l15;
        const int rb = mt * 16 + hshift;
#pragma unroll
        for (int v = 0; v < 8; ++v) {
          int row = rb + v;
          int gb = rowbase + row;
          float e = eps[((size_t)t * kB + gb) * kF + colr];
          float s = amu[v] + __expf(als[v]) * e;
          out[((size_t)gb * kS + t) * kF + colr] = s;
          lds_x[row * kF + colr] = (__bf16)s;
          if (t + 1 < kS)  // prefetch next step's eps (global_prefetch_b8)
            __builtin_prefetch(&eps[((size_t)(t + 1) * kB + gb) * kF + colr], 0, 3);
        }
      }
    }
    __syncthreads();
  }
}

// ---------------------------------------------------------------------------
extern "C" void kernel_launch(void* const* d_in, const int* in_sizes, int n_in,
                              void* d_out, int out_size, void* d_ws,
                              size_t ws_size, hipStream_t stream) {
  (void)in_sizes; (void)n_in; (void)out_size; (void)ws_size;
  const float* noise = (const float*)d_in[0];
  const float* eps = (const float*)d_in[1];
  const float* w1 = (const float*)d_in[2];
  const float* b1 = (const float*)d_in[3];
  const float* w2 = (const float*)d_in[4];
  const float* b2 = (const float*)d_in[5];
  const float* w3 = (const float*)d_in[6];
  const float* b3 = (const float*)d_in[7];
  const float* hw = (const float*)d_in[8];
  const float* hb = (const float*)d_in[9];
  const float* wih = (const float*)d_in[10];
  const float* bih = (const float*)d_in[11];
  const float* whh = (const float*)d_in[12];
  const float* bhh = (const float*)d_in[13];
  const float* outw = (const float*)d_in[14];
  const float* outb = (const float*)d_in[15];
  const float* zw = (const float*)d_in[16];
  const float* zb = (const float*)d_in[17];

  char* ws = (char*)d_ws;
  size_t off = 0;
  auto take = [&](size_t bytes) {
    void* p = ws + off;
    off = (off + bytes + 255) & ~(size_t)255;
    return p;
  };
  __bf16* wihb = (__bf16*)take(sizeof(short) * kG * kF);
  __bf16* whhb = (__bf16*)take(sizeof(short) * kG * kD);
  __bf16* outwb = (__bf16*)take(sizeof(short) * kF * kD);
  __bf16* zwb = (__bf16*)take(sizeof(short) * kD * kF);
  float* bcomb = (float*)take(sizeof(float) * kG);
  float* x0 = (float*)take(sizeof(float) * kB * kF);
  float* h0 = (float*)take(sizeof(float) * kB * kD);
  float* c0 = (float*)take(sizeof(float) * kB * kD);

  prep_kernel<<<(kG * kD + 255) / 256, 256, 0, stream>>>(
      wih, whh, outw, zw, bih, bhh, wihb, whhb, outwb, zwb, bcomb);

  preamble_kernel<<<kB, kF, 0, stream>>>(noise, w1, b1, w2, b2, w3, b3, hw, hb,
                                         x0, h0, c0);

  seq_kernel<<<kB / kRows, kThreads, 0, stream>>>(
      wihb, whhb, outwb, zwb, bcomb, outb, zb, x0, h0, c0, eps, (float*)d_out);
}
